// graphGAT_69810398429646
// MI455X (gfx1250) — matrix operand order repaired and checked by reading, not compiled, not run on hardware
//
#include <hip/hip_runtime.h>

// ---------------- dims (fixed by the reference) ----------------
constexpr int Dg   = 300;      // glove dim
constexpr int DP   = 320;      // padded glove dim (mult of 32)
constexpr int Tseq = 24;
constexpr int Hh   = 256;      // GRU hidden / z dim
constexpr int G3   = 768;      // 3*H
constexpr int NT   = 6000;
constexpr int NU   = 4000;
constexpr int Nn   = 10000;
constexpr int Ff   = 512;
constexpr int Zz   = 256;
constexpr int Bb   = 2000;
constexpr int H1   = 64;
constexpr int H2   = 100;
constexpr int H2P  = 128;      // padded H2 (mult of 64 -> uniform 4-tile strips)
constexpr int HEADS= 8;

typedef __attribute__((ext_vector_type(16))) _Float16 v16h;
typedef __attribute__((ext_vector_type(8)))  _Float16 v8h;
typedef __attribute__((ext_vector_type(8)))  float    v8f;

union frag16 { v16h v; v8h h[2]; };

__device__ __forceinline__ v8f wmma_f16(const frag16& a, const frag16& b, v8f c) {
    return __builtin_amdgcn_wmma_f32_16x16x32_f16(false, a.v, false, b.v,
                                                  (short)0, c, false, false);
}

// =====================================================================
// WMMA GEMM:  C[M,N] = A[M,K] * W[N,K]^T   (A,W f16 row-major, C f32)
// Requirements: M%16==0, N%64==0, K%32==0, 16B-aligned pointers.
// One wave computes a 16x64 strip (4 N-tiles): A fragment loaded once
// per k-step and reused by 4 back-to-back WMMAs. Straight-line loop,
// no EXEC manipulation. 8 waves per block.
// =====================================================================
__global__ void __launch_bounds__(256)
k_gemm_f16_wmma(const _Float16* __restrict__ A, const _Float16* __restrict__ W,
                float* __restrict__ C, int M, int N, int K)
{
    const int lane    = threadIdx.x & 31;
    const int wv      = threadIdx.x >> 5;
    const int stripsN = N >> 6;                     // strips of 4 N-tiles
    const int strip   = blockIdx.x * 8 + wv;
    if (strip >= (M >> 4) * stripsN) return;        // wave-uniform: EXEC all-ones
    const int tm  = strip / stripsN;
    const int tn0 = (strip % stripsN) * 4;
    const int r16 = lane & 15;
    const int hi  = lane >> 4;

    // A fragment: lane row = r16; halves e -> K = k + 16*hi + e
    const _Float16* ap = A + (size_t)(tm * 16 + r16) * K + hi * 16;
    // B fragments from W[N,K]: lane col = r16; halves e -> K = k + 16*hi + e
    const size_t bts = (size_t)16 * K;              // halfs between N-tiles
    const _Float16* bp0 = W + (size_t)(tn0 * 16 + r16) * K + hi * 16;
    const _Float16* bp1 = bp0 + bts;
    const _Float16* bp2 = bp0 + 2 * bts;
    const _Float16* bp3 = bp0 + 3 * bts;

    v8f a0 = {}, a1 = {}, a2 = {}, a3 = {};
#pragma unroll 2
    for (int k = 0; k < K; k += 32) {
        frag16 af, b0, b1, b2, b3;
        af.h[0] = *(const v8h*)(ap + k);
        af.h[1] = *(const v8h*)(ap + k + 8);
        __builtin_prefetch(ap + k + 128, 0, 1);     // -> global_prefetch_b8
        b0.h[0] = *(const v8h*)(bp0 + k);
        b0.h[1] = *(const v8h*)(bp0 + k + 8);
        b1.h[0] = *(const v8h*)(bp1 + k);
        b1.h[1] = *(const v8h*)(bp1 + k + 8);
        b2.h[0] = *(const v8h*)(bp2 + k);
        b2.h[1] = *(const v8h*)(bp2 + k + 8);
        b3.h[0] = *(const v8h*)(bp3 + k);
        b3.h[1] = *(const v8h*)(bp3 + k + 8);
        a0 = wmma_f16(af, b0, a0);
        a1 = wmma_f16(af, b1, a1);
        a2 = wmma_f16(af, b2, a2);
        a3 = wmma_f16(af, b3, a3);
    }
    // C layout: col = r16, row = vgpr + 8*hi
    float* cp = C + (size_t)(tm * 16 + hi * 8) * N + tn0 * 16 + r16;
#pragma unroll
    for (int v = 0; v < 8; ++v) {
        cp[(size_t)v * N]      = a0[v];
        cp[(size_t)v * N + 16] = a1[v];
        cp[(size_t)v * N + 32] = a2[v];
        cp[(size_t)v * N + 48] = a3[v];
    }
}

// ---------------- utility / conversion kernels ----------------
__global__ void k_zero_f32(float* p, long long n) {
    long long i = blockIdx.x * 256LL + threadIdx.x;
    if (i < n) p[i] = 0.f;
}
__global__ void k_fill_u32(unsigned* p, long long n, unsigned v) {
    long long i = blockIdx.x * 256LL + threadIdx.x;
    if (i < n) p[i] = v;
}
// dst[rows,dstK] f16 <- src[rows,srcK] f32, zero-pad cols
__global__ void k_cvt_pad(const float* __restrict__ s, _Float16* __restrict__ d,
                          int rows, int srcK, int dstK) {
    long long i = blockIdx.x * 256LL + threadIdx.x;
    long long tot = (long long)rows * dstK;
    if (i >= tot) return;
    int r = (int)(i / dstK), c = (int)(i % dstK);
    d[i] = (c < srcK) ? (_Float16)s[(long long)r * srcK + c] : (_Float16)0.f;
}
// dst[padN,K] f16 <- transpose(src[K,N] f32), zero-pad rows
__global__ void k_cvt_transpose_pad(const float* __restrict__ s, _Float16* __restrict__ d,
                                    int K, int N, int padN) {
    long long i = blockIdx.x * 256LL + threadIdx.x;
    long long tot = (long long)padN * K;
    if (i >= tot) return;
    int n = (int)(i / K), k = (int)(i % K);
    d[i] = (n < N) ? (_Float16)s[(long long)k * N + n] : (_Float16)0.f;
}
// embedding lookup for timestep t: xt[NT,DP] f16
__global__ void k_embed(const int* __restrict__ idx, const float* __restrict__ emb,
                        _Float16* __restrict__ xt, int t) {
    long long i = blockIdx.x * 256LL + threadIdx.x;
    long long tot = (long long)NT * DP;
    if (i >= tot) return;
    int r = (int)(i / DP), c = (int)(i % DP);
    xt[i] = (c < Dg) ? (_Float16)emb[(long long)idx[r * Tseq + t] * Dg + c]
                     : (_Float16)0.f;
}
__global__ void k_hinit(const float* __restrict__ h0in,
                        float* h0f, _Float16* h0h, float* h1f, _Float16* h1h) {
    long long i = blockIdx.x * 256LL + threadIdx.x;
    if (i >= (long long)NT * Hh) return;
    float a = h0in[i];
    float b = h0in[(long long)NT * Hh + i];
    h0f[i] = a; h0h[i] = (_Float16)a;
    h1f[i] = b; h1h[i] = (_Float16)b;
}
// fused GRU gates (torch GRU math), in-place h update + f16 copy
__global__ void k_gru_gates(const float* __restrict__ gi, const float* __restrict__ gh,
                            const float* __restrict__ b_ih, const float* __restrict__ b_hh,
                            float* __restrict__ h, _Float16* __restrict__ hh) {
    long long i = blockIdx.x * 256LL + threadIdx.x;
    if (i >= (long long)NT * Hh) return;
    int n = (int)(i / Hh), j = (int)(i % Hh);
    const float* gir = gi + (size_t)n * G3;
    const float* ghr = gh + (size_t)n * G3;
    float ir = gir[j]        + b_ih[j],        hr = ghr[j]        + b_hh[j];
    float iz = gir[Hh + j]   + b_ih[Hh + j],   hz = ghr[Hh + j]   + b_hh[Hh + j];
    float in_= gir[2*Hh + j] + b_ih[2*Hh + j], hn_= ghr[2*Hh + j] + b_hh[2*Hh + j];
    float r = 1.f / (1.f + __expf(-(ir + hr)));
    float z = 1.f / (1.f + __expf(-(iz + hz)));
    float nn = tanhf(in_ + r * hn_);
    float hv = (1.f - z) * nn + z * h[i];
    h[i] = hv; hh[i] = (_Float16)hv;
}
// VAE reparameterize + KL (block-reduced atomic)
__global__ void k_vae_post(const float* __restrict__ muraw, const float* __restrict__ lvraw,
                           const float* __restrict__ bmu, const float* __restrict__ blv,
                           const float* __restrict__ eps,
                           float* __restrict__ zf, _Float16* __restrict__ zh,
                           float* __restrict__ klout) {
    long long i = blockIdx.x * 256LL + threadIdx.x;
    long long tot = (long long)NU * Zz;
    float kl = 0.f;
    if (i < tot) {
        int j = (int)(i % Zz);
        float mu = muraw[i] + bmu[j];
        float lv = lvraw[i] + blv[j];
        float zv = mu + __expf(0.5f * lv) * eps[i];
        zf[i] = zv; zh[i] = (_Float16)zv;
        kl = -0.5f * (1.f + lv - mu * mu - __expf(lv)) * (1.f / NU);
    }
    __shared__ float red[256];
    red[threadIdx.x] = kl; __syncthreads();
    for (int s = 128; s > 0; s >>= 1) {
        if (threadIdx.x < s) red[threadIdx.x] += red[threadIdx.x + s];
        __syncthreads();
    }
    if (threadIdx.x == 0) atomicAdd(klout, red[0]);
}
__global__ void k_rec_loss(const float* __restrict__ recraw, const float* __restrict__ bdec,
                           const float* __restrict__ uf, float* __restrict__ recout) {
    long long i = blockIdx.x * 256LL + threadIdx.x;
    long long tot = (long long)NU * Ff;
    float v = 0.f;
    if (i < tot) {
        int j = (int)(i % Ff);
        float d = recraw[i] + bdec[j] - uf[i];
        v = d * d * (1.f / ((float)NU * (float)Ff));
    }
    __shared__ float red[256];
    red[threadIdx.x] = v; __syncthreads();
    for (int s = 128; s > 0; s >>= 1) {
        if (threadIdx.x < s) red[threadIdx.x] += red[threadIdx.x + s];
        __syncthreads();
    }
    if (threadIdx.x == 0) atomicAdd(recout, red[0]);
}
// x_in = concat(h1[:B], z, h1[B:]) as f16
__global__ void k_assemble_xin(const float* __restrict__ h1f, const float* __restrict__ zf,
                               _Float16* __restrict__ xin) {
    long long i = blockIdx.x * 256LL + threadIdx.x;
    if (i >= (long long)Nn * Hh) return;
    int n = (int)(i / Hh), j = (int)(i % Hh);
    float v;
    if (n < Bb)           v = h1f[(size_t)n * Hh + j];
    else if (n < Bb + NU) v = zf[(size_t)(n - Bb) * Hh + j];
    else                  v = h1f[(size_t)(n - NU) * Hh + j];
    xin[i] = (_Float16)v;
}
// per-node attention scores
__global__ void k_attn_scores(const float* __restrict__ xp, const float* __restrict__ asrc,
                              const float* __restrict__ adst, float* __restrict__ asn,
                              float* __restrict__ adn, int heads, int C, int stride) {
    int i = blockIdx.x * 256 + threadIdx.x;
    if (i >= Nn * heads) return;
    int n = i / heads, h = i % heads;
    const float* x = xp + (size_t)n * stride + h * C;
    float s = 0.f, d = 0.f;
    for (int c = 0; c < C; ++c) { s += x[c] * asrc[h * C + c]; d += x[c] * adst[h * C + c]; }
    asn[i] = s; adn[i] = d;
}
__device__ __forceinline__ unsigned f32_ord(float f) {
    unsigned u = __float_as_uint(f);
    return (u & 0x80000000u) ? ~u : (u | 0x80000000u);
}
__device__ __forceinline__ float ord_f32(unsigned m) {
    return (m & 0x80000000u) ? __uint_as_float(m & 0x7FFFFFFFu) : __uint_as_float(~m);
}
__global__ void k_edge_max(const int* __restrict__ ei, const float* __restrict__ asn,
                           const float* __restrict__ adn, float* __restrict__ e,
                           unsigned* __restrict__ nmax, int E, int heads) {
    long long i = blockIdx.x * 256LL + threadIdx.x;
    long long tot = (long long)(E + Nn) * heads;
    if (i >= tot) return;
    long long eid = i / heads; int h = (int)(i % heads);
    int s, d;
    if (eid < E) { s = ei[eid]; d = ei[E + eid]; } else { s = d = (int)(eid - E); }
    float v = asn[s * heads + h] + adn[d * heads + h];
    v = (v >= 0.f) ? v : 0.2f * v;          // leaky_relu(0.2)
    e[i] = v;
    atomicMax(&nmax[d * heads + h], f32_ord(v));
}
__global__ void k_edge_expsum(const int* __restrict__ ei, float* __restrict__ e,
                              const unsigned* __restrict__ nmax, float* __restrict__ den,
                              int E, int heads) {
    long long i = blockIdx.x * 256LL + threadIdx.x;
    long long tot = (long long)(E + Nn) * heads;
    if (i >= tot) return;
    long long eid = i / heads; int h = (int)(i % heads);
    int d = (eid < E) ? ei[E + eid] : (int)(eid - E);
    float ee = __expf(e[i] - ord_f32(nmax[d * heads + h]));
    e[i] = ee;
    atomicAdd(&den[d * heads + h], ee);
}
__global__ void k_edge_agg(const int* __restrict__ ei, const float* __restrict__ e,
                           const float* __restrict__ den, const float* __restrict__ xp,
                           float* __restrict__ agg, int E, int heads, int C,
                           int xpStride, int aggStride) {
    long long i = blockIdx.x * 256LL + threadIdx.x;
    long long tot = (long long)(E + Nn) * heads * C;
    if (i >= tot) return;
    int c = (int)(i % C);
    long long t = i / C;
    int h = (int)(t % heads);
    long long eid = t / heads;
    int s, d;
    if (eid < E) { s = ei[eid]; d = ei[E + eid]; } else { s = d = (int)(eid - E); }
    float alpha = e[eid * heads + h] / den[d * heads + h];
    atomicAdd(&agg[(size_t)d * aggStride + h * C + c],
              xp[(size_t)s * xpStride + h * C + c] * alpha);
}
__global__ void k_gat1_epi(const float* __restrict__ agg, const float* __restrict__ b,
                           _Float16* __restrict__ x1h) {
    long long i = blockIdx.x * 256LL + threadIdx.x;
    if (i >= (long long)Nn * (HEADS * H1)) return;
    int j = (int)(i % (HEADS * H1));
    x1h[i] = (_Float16)(agg[i] + b[j]);
}
__global__ void k_final(const float* __restrict__ agg2, const float* __restrict__ b2,
                        float* __restrict__ out) {
    long long i = blockIdx.x * 256LL + threadIdx.x;
    if (i >= (long long)Nn * H2) return;
    int j = (int)(i % H2);
    out[i] = agg2[i] + b2[j];
}

// =====================================================================
extern "C" void kernel_launch(void* const* d_in, const int* in_sizes, int n_in,
                              void* d_out, int out_size, void* d_ws, size_t ws_size,
                              hipStream_t stream)
{
    const float* user_feats = (const float*)d_in[1];
    const int*   node_idx   = (const int*)d_in[2];
    const int*   edge_index = (const int*)d_in[3];
    const float* tweet_emb  = (const float*)d_in[4];
    const float* w_ih0 = (const float*)d_in[5],  *w_hh0 = (const float*)d_in[6];
    const float* b_ih0 = (const float*)d_in[7],  *b_hh0 = (const float*)d_in[8];
    const float* w_ih1 = (const float*)d_in[9],  *w_hh1 = (const float*)d_in[10];
    const float* b_ih1 = (const float*)d_in[11], *b_hh1 = (const float*)d_in[12];
    const float* h0in  = (const float*)d_in[13];
    const float* w_mu  = (const float*)d_in[14], *b_mu  = (const float*)d_in[15];
    const float* w_lv  = (const float*)d_in[16], *b_lv  = (const float*)d_in[17];
    const float* w_dec = (const float*)d_in[18], *b_dec = (const float*)d_in[19];
    const float* vae_eps = (const float*)d_in[20];
    const float* w1    = (const float*)d_in[21];
    const float* a_src1= (const float*)d_in[22], *a_dst1= (const float*)d_in[23];
    const float* b1    = (const float*)d_in[24];
    const float* w2    = (const float*)d_in[25];
    const float* a_src2= (const float*)d_in[26], *a_dst2= (const float*)d_in[27];
    const float* b2    = (const float*)d_in[28];
    float* outF = (float*)d_out;
    const int E = in_sizes[3] / 2;           // 80000
    const int EL = E + Nn;                   // with self loops

    // ---- workspace bump allocator (256B aligned) ----
    size_t off = 0;
    auto alloc = [&](size_t bytes) -> void* {
        off = (off + 255) & ~(size_t)255;
        void* p = (char*)d_ws + off;
        off += bytes;
        return p;
    };
    _Float16* wih0h = (_Float16*)alloc((size_t)G3 * DP * 2);
    _Float16* whh0h = (_Float16*)alloc((size_t)G3 * Hh * 2);
    _Float16* wih1h = (_Float16*)alloc((size_t)G3 * Hh * 2);
    _Float16* whh1h = (_Float16*)alloc((size_t)G3 * Hh * 2);
    _Float16* wmuh  = (_Float16*)alloc((size_t)Zz * Ff * 2);
    _Float16* wlvh  = (_Float16*)alloc((size_t)Zz * Ff * 2);
    _Float16* wdech = (_Float16*)alloc((size_t)Ff * Zz * 2);
    _Float16* w1th  = (_Float16*)alloc((size_t)(HEADS*H1) * Hh * 2);
    _Float16* w2th  = (_Float16*)alloc((size_t)H2P * (HEADS*H1) * 2);
    _Float16* ufh   = (_Float16*)alloc((size_t)NU * Ff * 2);
    _Float16* xth   = (_Float16*)alloc((size_t)NT * DP * 2);
    float*    gi    = (float*)alloc((size_t)NT * G3 * 4);
    float*    gh    = (float*)alloc((size_t)NT * G3 * 4);
    float*    h0f   = (float*)alloc((size_t)NT * Hh * 4);
    _Float16* h0h   = (_Float16*)alloc((size_t)NT * Hh * 2);
    float*    h1f   = (float*)alloc((size_t)NT * Hh * 4);
    _Float16* h1h   = (_Float16*)alloc((size_t)NT * Hh * 2);
    float*    muB   = (float*)alloc((size_t)NU * Zz * 4);
    float*    lvB   = (float*)alloc((size_t)NU * Zz * 4);
    float*    zf    = (float*)alloc((size_t)NU * Zz * 4);
    _Float16* zh    = (_Float16*)alloc((size_t)NU * Zz * 2);
    float*    recr  = (float*)alloc((size_t)NU * Ff * 4);
    _Float16* xinh  = (_Float16*)alloc((size_t)Nn * Hh * 2);
    float*    xp1   = (float*)alloc((size_t)Nn * HEADS * H1 * 4);
    _Float16* x1h   = (_Float16*)alloc((size_t)Nn * HEADS * H1 * 2);
    float*    as1   = (float*)alloc((size_t)Nn * HEADS * 4);
    float*    ad1   = (float*)alloc((size_t)Nn * HEADS * 4);
    unsigned* nmax1 = (unsigned*)alloc((size_t)Nn * HEADS * 4);
    float*    den1  = (float*)alloc((size_t)Nn * HEADS * 4);
    float*    ea1   = (float*)alloc((size_t)EL * HEADS * 4);
    float*    agg1  = (float*)alloc((size_t)Nn * HEADS * H1 * 4);
    float*    xp2   = (float*)alloc((size_t)Nn * H2P * 4);
    float*    as2   = (float*)alloc((size_t)Nn * 4);
    float*    ad2   = (float*)alloc((size_t)Nn * 4);
    unsigned* nmax2 = (unsigned*)alloc((size_t)Nn * 4);
    float*    den2  = (float*)alloc((size_t)Nn * 4);
    float*    ea2   = (float*)alloc((size_t)EL * 4);
    float*    agg2  = (float*)alloc((size_t)Nn * H2 * 4);
    (void)ws_size;

    auto GB = [](long long n) { return dim3((unsigned)((n + 255) / 256)); };
    auto gemm = [&](const _Float16* A, const _Float16* W, float* C, int M, int N, int K) {
        long long strips = (long long)(M / 16) * (N / 64);
        k_gemm_f16_wmma<<<dim3((unsigned)((strips + 7) / 8)), 256, 0, stream>>>(A, W, C, M, N, K);
    };

    // ---- weight conversion (once per call; deterministic) ----
    k_cvt_pad<<<GB((long long)G3*DP), 256, 0, stream>>>(w_ih0, wih0h, G3, Dg, DP);
    k_cvt_pad<<<GB((long long)G3*Hh), 256, 0, stream>>>(w_hh0, whh0h, G3, Hh, Hh);
    k_cvt_pad<<<GB((long long)G3*Hh), 256, 0, stream>>>(w_ih1, wih1h, G3, Hh, Hh);
    k_cvt_pad<<<GB((long long)G3*Hh), 256, 0, stream>>>(w_hh1, whh1h, G3, Hh, Hh);
    k_cvt_pad<<<GB((long long)Zz*Ff), 256, 0, stream>>>(w_mu,  wmuh,  Zz, Ff, Ff);
    k_cvt_pad<<<GB((long long)Zz*Ff), 256, 0, stream>>>(w_lv,  wlvh,  Zz, Ff, Ff);
    k_cvt_pad<<<GB((long long)Ff*Zz), 256, 0, stream>>>(w_dec, wdech, Ff, Zz, Zz);
    k_cvt_pad<<<GB((long long)NU*Ff), 256, 0, stream>>>(user_feats, ufh, NU, Ff, Ff);
    k_cvt_transpose_pad<<<GB((long long)(HEADS*H1)*Hh), 256, 0, stream>>>(w1, w1th, Hh, HEADS*H1, HEADS*H1);
    k_cvt_transpose_pad<<<GB((long long)H2P*(HEADS*H1)), 256, 0, stream>>>(w2, w2th, HEADS*H1, H2, H2P);

    // ---- init accumulators ----
    k_zero_f32<<<GB(2), 256, 0, stream>>>(outF + (out_size - 2), 2);
    k_zero_f32<<<GB((long long)Nn*HEADS*H1), 256, 0, stream>>>(agg1, (long long)Nn*HEADS*H1);
    k_zero_f32<<<GB((long long)Nn*HEADS), 256, 0, stream>>>(den1, (long long)Nn*HEADS);
    k_fill_u32<<<GB((long long)Nn*HEADS), 256, 0, stream>>>(nmax1, (long long)Nn*HEADS, 0u);
    k_zero_f32<<<GB((long long)Nn*H2), 256, 0, stream>>>(agg2, (long long)Nn*H2);
    k_zero_f32<<<GB(Nn), 256, 0, stream>>>(den2, Nn);
    k_fill_u32<<<GB(Nn), 256, 0, stream>>>(nmax2, Nn, 0u);

    // ---- GRU: fused 2-layer scan over T=24 ----
    k_hinit<<<GB((long long)NT*Hh), 256, 0, stream>>>(h0in, h0f, h0h, h1f, h1h);
    for (int t = 0; t < Tseq; ++t) {
        k_embed<<<GB((long long)NT*DP), 256, 0, stream>>>(node_idx, tweet_emb, xth, t);
        gemm(xth, wih0h, gi, NT, G3, DP);                 // x_t @ w_ih0^T
        gemm(h0h, whh0h, gh, NT, G3, Hh);                 // h0  @ w_hh0^T
        k_gru_gates<<<GB((long long)NT*Hh), 256, 0, stream>>>(gi, gh, b_ih0, b_hh0, h0f, h0h);
        gemm(h0h, wih1h, gi, NT, G3, Hh);                 // out0_t @ w_ih1^T
        gemm(h1h, whh1h, gh, NT, G3, Hh);                 // h1  @ w_hh1^T
        k_gru_gates<<<GB((long long)NT*Hh), 256, 0, stream>>>(gi, gh, b_ih1, b_hh1, h1f, h1h);
    }

    // ---- VAE ----
    gemm(ufh, wmuh, muB, NU, Zz, Ff);
    gemm(ufh, wlvh, lvB, NU, Zz, Ff);
    k_vae_post<<<GB((long long)NU*Zz), 256, 0, stream>>>(muB, lvB, b_mu, b_lv, vae_eps,
                                                         zf, zh, outF + (out_size - 2));
    gemm(zh, wdech, recr, NU, Ff, Zz);
    k_rec_loss<<<GB((long long)NU*Ff), 256, 0, stream>>>(recr, b_dec, user_feats,
                                                         outF + (out_size - 1));

    // ---- GAT layer 1 (8 heads x 64, concat) ----
    k_assemble_xin<<<GB((long long)Nn*Hh), 256, 0, stream>>>(h1f, zf, xinh);
    gemm(xinh, w1th, xp1, Nn, HEADS*H1, Hh);
    k_attn_scores<<<GB((long long)Nn*HEADS), 256, 0, stream>>>(xp1, a_src1, a_dst1, as1, ad1,
                                                               HEADS, H1, HEADS*H1);
    k_edge_max<<<GB((long long)EL*HEADS), 256, 0, stream>>>(edge_index, as1, ad1, ea1, nmax1, E, HEADS);
    k_edge_expsum<<<GB((long long)EL*HEADS), 256, 0, stream>>>(edge_index, ea1, nmax1, den1, E, HEADS);
    k_edge_agg<<<GB((long long)EL*HEADS*H1), 256, 0, stream>>>(edge_index, ea1, den1, xp1, agg1,
                                                               E, HEADS, H1, HEADS*H1, HEADS*H1);
    k_gat1_epi<<<GB((long long)Nn*HEADS*H1), 256, 0, stream>>>(agg1, b1, x1h);

    // ---- GAT layer 2 (1 head x 100, mean==identity) ----
    gemm(x1h, w2th, xp2, Nn, H2P, HEADS*H1);
    k_attn_scores<<<GB((long long)Nn), 256, 0, stream>>>(xp2, a_src2, a_dst2, as2, ad2, 1, H2, H2P);
    k_edge_max<<<GB((long long)EL), 256, 0, stream>>>(edge_index, as2, ad2, ea2, nmax2, E, 1);
    k_edge_expsum<<<GB((long long)EL), 256, 0, stream>>>(edge_index, ea2, nmax2, den2, E, 1);
    k_edge_agg<<<GB((long long)EL*H2), 256, 0, stream>>>(edge_index, ea2, den2, xp2, agg2,
                                                         E, 1, H2, H2P, H2);
    k_final<<<GB((long long)Nn*H2), 256, 0, stream>>>(agg2, b2, outF);
}
